// F0Predictor_48644799594511
// MI455X (gfx1250) — matrix operand order repaired and checked by reading, not compiled, not run on hardware
//
#include <hip/hip_runtime.h>
#include <hip/hip_bf16.h>

typedef __bf16 bf16_t;
typedef __attribute__((ext_vector_type(16))) __bf16 v16bf;
typedef __attribute__((ext_vector_type(8)))  float  v8f;
typedef __attribute__((ext_vector_type(4)))  unsigned int v4u;
typedef __attribute__((ext_vector_type(8)))  unsigned int v8u;

__device__ __forceinline__ float sigmoidf_(float x) { return 1.0f / (1.0f + __expf(-x)); }

#define WMMA_BF16(A_, B_, C_) \
    __builtin_amdgcn_wmma_f32_16x16x32_bf16(false, (A_), false, (B_), (short)0, (C_), false, false)

// ---------------------------------------------------------------------------
// Tensor Data Mover: DMA a 2-D tile (tile_d1 rows x tile_d0 elements, 2-byte
// elements) from global [rows x K] (row stride = stride_elems) into LDS at
// lds_off, stored linearly row-major. D# per CDNA5 ISA 8.3/8.4; 2-D tensor ->
// descriptor groups 2/3 omitted. TDM ignores EXEC: call from ONE wave only.
// Completion tracked by TENSORcnt (in-order per wave).
// ---------------------------------------------------------------------------
__device__ __forceinline__ void tdm_load_2d(unsigned lds_off, const void* gaddr,
                                            unsigned tensor_d0, unsigned tensor_d1,
                                            unsigned tile_d0, unsigned tile_d1,
                                            unsigned stride_elems) {
    unsigned long long ga = (unsigned long long)gaddr;
    v4u g0;
    g0[0] = 1u;                                   // count=1 (valid user D#)
    g0[1] = lds_off;                              // lds_addr (bytes)
    g0[2] = (unsigned)ga;                         // global_addr[31:0]
    g0[3] = (unsigned)(ga >> 32) | (2u << 30);    // global_addr[56:32] | type=2
    v8u g1;
    g1[0] = 1u << 16;                             // data_size=1 (2B); no mask/pad
    g1[1] = (tensor_d0 & 0xFFFFu) << 16;          // tensor_dim0[15:0] @ bit48
    g1[2] = (tensor_d0 >> 16) | ((tensor_d1 & 0xFFFFu) << 16);  // d0[31:16], d1[15:0]
    g1[3] = (tensor_d1 >> 16) | (tile_d0 << 16);  // d1[31:16], tile_dim0
    g1[4] = tile_d1;                              // tile_dim1 (tile_dim2 = 0)
    g1[5] = stride_elems;                         // tensor_dim0_stride[31:0]
    g1[6] = 0;                                    // stride hi / dim1_stride lo
    g1[7] = 0;
    asm volatile("tensor_load_to_lds %0, %1" :: "s"(g0), "s"(g1) : "memory");
}

// ---------------------------------------------------------------------------
// fp32 -> bf16 conversion
// ---------------------------------------------------------------------------
__global__ void f32_to_bf16_kernel(const float* __restrict__ src,
                                   bf16_t* __restrict__ dst, int n) {
    int i = blockIdx.x * blockDim.x + threadIdx.x;
    if (i < n) dst[i] = (bf16_t)src[i];
}

// ===========================================================================
// WMMA fragment layouts (ISA 7.12.2, 16-bit operands, wave32):
//  A frag (16x32, row M=lane%16): lane<16 holds K{0-7,16-23}, lane>=16 K{8-15,24-31}
//  B frag (32x16, col N=lane%16): as A; gemm kernel: A,B lane<16 K0-15, lane>=16 K16-31
// From a row-major LDS tile, A frag = two contiguous 16B runs, B frag = one
// contiguous 32B run -> 2 x ds_load_b128 each.
// ===========================================================================

// ---------------------------------------------------------------------------
// Implicit-GEMM conv1d (KW=31, stride=4, pad=15) + bias + ReLU, bf16 in/out.
// Gathered boundary-checked staging into fragment-order (swizzled) LDS.
// Workgroup tile 128(M) x 64(N); 8 waves, each 32x32 via 2x2 WMMA; K-step 32.
// ---------------------------------------------------------------------------
__global__ __launch_bounds__(256)
void conv_wmma_kernel(const bf16_t* __restrict__ in, const bf16_t* __restrict__ w,
                      const float* __restrict__ bias, bf16_t* __restrict__ out,
                      int B, int Cin, int Lin, int Cout, int Lout, int K, int Kpad) {
    const int KW = 31, STRIDE = 4, PADL = 15;
    __shared__ alignas(32) bf16_t As[8][32][16];  // 8 M-blocks of 16 rows
    __shared__ alignas(32) bf16_t Ws[4][32][16];  // 4 N-blocks of 16 cols

    const int tid  = threadIdx.x;
    const int lane = tid & 31;
    const int wave = tid >> 5;
    const int wm   = wave & 3;   // 0..3 (M, 32 rows each)
    const int wn   = wave >> 2;  // 0..1 (N, 32 cols each)
    const long m0  = (long)blockIdx.x * 128;
    const int  n0  = blockIdx.y * 64;

    v8f acc[2][2] = {};

    for (int k0 = 0; k0 < Kpad; k0 += 32) {
        for (int idx = tid; idx < 128 * 32; idx += 256) {
            int ml = idx >> 5, kk = idx & 31;
            int k  = k0 + kk;
            bf16_t v = (bf16_t)0.0f;
            if (k < K) {
                int  ci = k / KW;
                int  kw = k - ci * KW;
                long m  = m0 + ml;
                int  b  = (int)(m / Lout);
                int  lo = (int)(m % Lout);
                int  li = lo * STRIDE - PADL + kw;
                if (li >= 0 && li < Lin)
                    v = in[((long)b * Cin + ci) * Lin + li];
            }
            As[ml >> 4][(ml & 15) + 16 * ((kk >> 3) & 1)][(kk & 7) + 8 * ((kk >> 4) & 1)] = v;
        }
        for (int idx = tid; idx < 64 * 32; idx += 256) {
            int nl = idx >> 5, kk = idx & 31;
            int k  = k0 + kk;
            bf16_t v = (bf16_t)0.0f;
            if (k < K) v = w[(long)(n0 + nl) * K + k];
            Ws[nl >> 4][(nl & 15) + 16 * ((kk >> 3) & 1)][(kk & 7) + 8 * ((kk >> 4) & 1)] = v;
        }
        __syncthreads();

        v16bf a0 = *(const v16bf*)As[wm * 2 + 0][lane];
        v16bf a1 = *(const v16bf*)As[wm * 2 + 1][lane];
        v16bf b0 = *(const v16bf*)Ws[wn * 2 + 0][lane];
        v16bf b1 = *(const v16bf*)Ws[wn * 2 + 1][lane];
        acc[0][0] = WMMA_BF16(a0, b0, acc[0][0]);
        acc[0][1] = WMMA_BF16(a0, b1, acc[0][1]);
        acc[1][0] = WMMA_BF16(a1, b0, acc[1][0]);
        acc[1][1] = WMMA_BF16(a1, b1, acc[1][1]);
        __syncthreads();
    }

#pragma unroll
    for (int j = 0; j < 2; ++j) {
        int nc = n0 + wn * 32 + j * 16 + (lane & 15);
        float bv = bias[nc];
#pragma unroll
        for (int i = 0; i < 2; ++i) {
#pragma unroll
            for (int r = 0; r < 8; ++r) {
                long m  = m0 + wm * 32 + i * 16 + r + ((lane >= 16) ? 8 : 0);
                int  b  = (int)(m / Lout);
                int  lo = (int)(m % Lout);
                float v = acc[i][j][r] + bv;
                v = v > 0.0f ? v : 0.0f;  // ReLU
                out[((long)b * Cout + nc) * Lout + lo] = (bf16_t)v;
            }
        }
    }
}

// ---------------------------------------------------------------------------
// Generic C = A @ W^T (+bias). A:[M,K] bf16 row-major, W:[N,K] bf16.
// Tiles are DMA'd into LDS by the Tensor Data Mover (wave 0 issues; TENSORcnt
// pipelined double-buffering overlaps DMA of slab t+1 with WMMA on slab t).
// Workgroup tile 128(M) x 64(N); K-tile 64 -> 8 WMMAs/wave/iter.
// M % 128 == 0, N % 64 == 0, K % 64 == 0 guaranteed by callers.
// ---------------------------------------------------------------------------
__global__ __launch_bounds__(256)
void gemm_wmma_kernel(const bf16_t* __restrict__ A, const bf16_t* __restrict__ W,
                      const float* __restrict__ bias, float* __restrict__ Cf,
                      bf16_t* __restrict__ Cb, int M, int N, int K) {
    __shared__ alignas(32) bf16_t At[2][128][64];  // 32KB, row-major [m][k]
    __shared__ alignas(32) bf16_t Bt[2][64][64];   // 16KB, row-major [n][k]

    const int tid  = threadIdx.x;
    const int lane = tid & 31;
    const int wave = tid >> 5;
    const int wm   = wave & 3;
    const int wn   = wave >> 2;
    const long m0  = (long)blockIdx.x * 128;
    const int  n0  = blockIdx.y * 64;

    const unsigned ldsA[2] = {(unsigned)(unsigned long long)&At[0][0][0],
                              (unsigned)(unsigned long long)&At[1][0][0]};
    const unsigned ldsB[2] = {(unsigned)(unsigned long long)&Bt[0][0][0],
                              (unsigned)(unsigned long long)&Bt[1][0][0]};

    v8f acc[2][2] = {};
    int buf = 0;

    if (wave == 0) {  // prologue: DMA first K-slab into buffer 0
        tdm_load_2d(ldsA[0], A + m0 * (long)K, K, (unsigned)M, 64, 128, K);
        tdm_load_2d(ldsB[0], W + (long)n0 * K, K, (unsigned)N, 64, 64, K);
    }

    for (int k0 = 0; k0 < K; k0 += 64) {
        if (wave == 0) {
            if (k0 + 64 < K) {  // DMA next slab into the other buffer
                tdm_load_2d(ldsA[buf ^ 1], A + m0 * (long)K + k0 + 64, K, (unsigned)M, 64, 128, K);
                tdm_load_2d(ldsB[buf ^ 1], W + (long)n0 * K + k0 + 64, K, (unsigned)N, 64, 64, K);
                __builtin_amdgcn_s_wait_tensorcnt(2);  // current slab's 2 loads done
            } else {
                __builtin_amdgcn_s_wait_tensorcnt(0);
            }
        }
        __syncthreads();  // buffer `buf` is filled

        const int r16  = lane & 15;
        const int h8   = (lane >> 4) << 4;   // A: 0 or 16
        const int kh16 = (lane >> 4) << 4;   // B: 0 or 16
#pragma unroll
        for (int s = 0; s < 2; ++s) {
            const bf16_t* ar0 = &At[buf][wm * 32 + r16][s * 32];
            const bf16_t* ar1 = &At[buf][wm * 32 + 16 + r16][s * 32];
            v16bf a0, a1, b0, b1;
            *(uint4*)&a0       = *(const uint4*)(ar0 + h8);
            *((uint4*)&a0 + 1) = *(const uint4*)(ar0 + 8 + h8);
            *(uint4*)&a1       = *(const uint4*)(ar1 + h8);
            *((uint4*)&a1 + 1) = *(const uint4*)(ar1 + 8 + h8);
            b0 = *(const v16bf*)&Bt[buf][wn * 32 + r16][s * 32 + kh16];
            b1 = *(const v16bf*)&Bt[buf][wn * 32 + 16 + r16][s * 32 + kh16];
            acc[0][0] = WMMA_BF16(a0, b0, acc[0][0]);
            acc[0][1] = WMMA_BF16(a0, b1, acc[0][1]);
            acc[1][0] = WMMA_BF16(a1, b0, acc[1][0]);
            acc[1][1] = WMMA_BF16(a1, b1, acc[1][1]);
        }
        __syncthreads();  // all waves done with `buf` before TDM refills it
        buf ^= 1;
    }

#pragma unroll
    for (int j = 0; j < 2; ++j) {
        int nc = n0 + wn * 32 + j * 16 + (lane & 15);
        float bv = bias ? bias[nc] : 0.0f;
#pragma unroll
        for (int i = 0; i < 2; ++i) {
#pragma unroll
            for (int r = 0; r < 8; ++r) {
                long m  = m0 + wm * 32 + i * 16 + r + ((lane >= 16) ? 8 : 0);
                float v = acc[i][j][r] + bv;
                if (Cf) Cf[m * (long)N + nc] = v;
                if (Cb) Cb[m * (long)N + nc] = (bf16_t)v;
            }
        }
    }
}

// ---------------------------------------------------------------------------
// hflat[b, l*1024 + c] = conv4out[b, c, l]   (conv4out: [B,1024,8], bf16)
// ---------------------------------------------------------------------------
__global__ void repack_hflat_kernel(const bf16_t* __restrict__ a4,
                                    bf16_t* __restrict__ hflat, int B) {
    int i = blockIdx.x * blockDim.x + threadIdx.x;
    if (i >= B * 8192) return;
    int b = i >> 13, r = i & 8191;
    int l = r >> 10, c = r & 1023;
    hflat[i] = a4[((long)b * 1024 + c) * 8 + l];
}

// ---------------------------------------------------------------------------
// Rank-2 input-path precompute:
//   u0 = w_ih @ emb_w[:,0]; u1 = w_ih @ emb_w[:,1]; u2 = w_ih @ emb_b;
//   bvec = b_ih + b_hh.                     (length 4*RNN = 2048, fp32)
// ---------------------------------------------------------------------------
__global__ void prep_u_kernel(const float* __restrict__ w_ih,
                              const float* __restrict__ emb_w,
                              const float* __restrict__ emb_b,
                              const float* __restrict__ b_ih,
                              const float* __restrict__ b_hh,
                              float* __restrict__ u0, float* __restrict__ u1,
                              float* __restrict__ u2, float* __restrict__ bvec) {
    int j = blockIdx.x * blockDim.x + threadIdx.x;
    if (j >= 2048) return;
    const float* row = w_ih + (long)j * 512;
    float s0 = 0.f, s1 = 0.f, s2 = 0.f;
    for (int k = 0; k < 512; ++k) {
        float wv = row[k];
        s0 += wv * emb_w[2 * k + 0];
        s1 += wv * emb_w[2 * k + 1];
        s2 += wv * emb_b[k];
    }
    u0[j] = s0; u1[j] = s1; u2[j] = s2;
    bvec[j] = b_ih[j] + b_hh[j];
}

// ---------------------------------------------------------------------------
// Fused LSTM cell + output heads for one time step.
// ---------------------------------------------------------------------------
__global__ __launch_bounds__(512)
void lstm_cell_kernel(const float* __restrict__ gates, const float* __restrict__ u0,
                      const float* __restrict__ u1, const float* __restrict__ u2,
                      const float* __restrict__ bvec, float* __restrict__ c,
                      bf16_t* __restrict__ hb, const float* __restrict__ lf0_w,
                      const float* __restrict__ lf0_b, const float* __restrict__ uv_w,
                      const float* __restrict__ uv_b, float* __restrict__ sc,
                      float* __restrict__ out, int t, int num_steps, int has_x) {
    __shared__ float r1[512];
    __shared__ float r2[512];
    const int b = blockIdx.x;
    const int j = threadIdx.x;

    float lf0p = 0.f, uvp = 0.f;
    if (has_x) { lf0p = sc[2 * b]; uvp = sc[2 * b + 1]; }

    const float* g = gates + (long)b * 2048;
    float pre[4];
#pragma unroll
    for (int q = 0; q < 4; ++q) {
        int idx = q * 512 + j;
        float v = g[idx] + bvec[idx];
        if (has_x) v += lf0p * u0[idx] + uvp * u1[idx] + u2[idx];
        pre[q] = v;
    }
    float iv = sigmoidf_(pre[0]);
    float fv = sigmoidf_(pre[1]);
    float gv = tanhf(pre[2]);
    float ov = sigmoidf_(pre[3]);

    float cn = fv * c[b * 512 + j] + iv * gv;
    c[b * 512 + j] = cn;
    float hn = ov * tanhf(cn);
    hb[b * 512 + j] = (bf16_t)hn;

    r1[j] = hn * lf0_w[j];
    r2[j] = hn * uv_w[j];
    __syncthreads();
    for (int s = 256; s > 0; s >>= 1) {
        if (j < s) { r1[j] += r1[j + s]; r2[j] += r2[j + s]; }
        __syncthreads();
    }
    if (j == 0) {
        float lf0 = r1[0] + lf0_b[0];
        float uv  = sigmoidf_(r2[0] + uv_b[0]);
        sc[2 * b]     = lf0;
        sc[2 * b + 1] = uv;
        out[((long)b * num_steps + t) * 2 + 0] = lf0;
        out[((long)b * num_steps + t) * 2 + 1] = uv;
    }
}

// ---------------------------------------------------------------------------
// Host-side orchestration
// ---------------------------------------------------------------------------
extern "C" void kernel_launch(void* const* d_in, const int* in_sizes, int n_in,
                              void* d_out, int out_size, void* d_ws, size_t ws_size,
                              hipStream_t stream) {
    const float* x     = (const float*)d_in[0];
    const float* cwp[5] = {(const float*)d_in[2], (const float*)d_in[4],
                           (const float*)d_in[6], (const float*)d_in[8],
                           (const float*)d_in[10]};
    const float* cbp[5] = {(const float*)d_in[3], (const float*)d_in[5],
                           (const float*)d_in[7], (const float*)d_in[9],
                           (const float*)d_in[11]};
    const float* ph_w  = (const float*)d_in[12];
    const float* ph_b  = (const float*)d_in[13];
    const float* pc_w  = (const float*)d_in[14];
    const float* pc_b  = (const float*)d_in[15];
    const float* w_ih  = (const float*)d_in[16];
    const float* w_hh  = (const float*)d_in[17];
    const float* b_ih  = (const float*)d_in[18];
    const float* b_hh  = (const float*)d_in[19];
    const float* lf0_w = (const float*)d_in[20];
    const float* lf0_b = (const float*)d_in[21];
    const float* uv_w  = (const float*)d_in[22];
    const float* uv_b  = (const float*)d_in[23];
    const float* emb_w = (const float*)d_in[24];
    const float* emb_b = (const float*)d_in[25];

    const int B = in_sizes[0] / 8192;          // 128
    const int num_steps = out_size / (B * 2);  // 200 (derived, capture-safe)

    // --- workspace bump allocator (256B aligned) ---
    char* ws = (char*)d_ws;
    size_t off = 0;
    auto alloc = [&](size_t bytes) -> char* {
        char* p = ws + off;
        off = (off + bytes + 255) & ~(size_t)255;
        return p;
    };
    const long PING_E = 16777216;  // max bf16 activation (layer0 out)
    bf16_t* ping  = (bf16_t*)alloc(PING_E * 2);
    bf16_t* pong  = (bf16_t*)alloc(PING_E * 2);
    const long wcE[5] = {64L * 31, 128L * 64 * 31, 256L * 128 * 31,
                         512L * 256 * 31, 1024L * 512 * 31};
    bf16_t* wc[5];
    for (int i = 0; i < 5; ++i) wc[i] = (bf16_t*)alloc(wcE[i] * 2);
    bf16_t* phb   = (bf16_t*)alloc(512L * 8192 * 2);
    bf16_t* pcb   = (bf16_t*)alloc(512L * 8192 * 2);
    bf16_t* whhb  = (bf16_t*)alloc(2048L * 512 * 2);
    bf16_t* hb    = (bf16_t*)alloc((long)B * 512 * 2);
    float*  cbuf  = (float*)alloc((long)B * 512 * 4);
    float*  gates = (float*)alloc((long)B * 2048 * 4);
    float*  u0    = (float*)alloc(2048 * 4);
    float*  u1    = (float*)alloc(2048 * 4);
    float*  u2    = (float*)alloc(2048 * 4);
    float*  bvec  = (float*)alloc(2048 * 4);
    float*  sc    = (float*)alloc((long)B * 2 * 4);

    auto cvt = [&](const float* s, bf16_t* d, long n) {
        f32_to_bf16_kernel<<<(int)((n + 255) / 256), 256, 0, stream>>>(s, d, (int)n);
    };

    // --- weight & input conversions to bf16 ---
    cvt(x, ping, (long)B * 8192);              // x -> ping
    for (int i = 0; i < 5; ++i) cvt(cwp[i], wc[i], wcE[i]);
    cvt(ph_w, phb, 512L * 8192);
    cvt(pc_w, pcb, 512L * 8192);
    cvt(w_hh, whhb, 2048L * 512);

    // --- conv stack (ping/pong) ---
    struct { int Cin, Lin, Cout, Lout; } L[5] = {
        {1, 8192, 64, 2048}, {64, 2048, 128, 512}, {128, 512, 256, 128},
        {256, 128, 512, 32}, {512, 32, 1024, 8}};
    bf16_t* cur = ping;
    bf16_t* nxt = pong;
    for (int i = 0; i < 5; ++i) {
        int M = B * L[i].Lout;
        int K = L[i].Cin * 31;
        int Kpad = ((K + 31) / 32) * 32;
        dim3 grid(M / 128, L[i].Cout / 64);
        conv_wmma_kernel<<<grid, 256, 0, stream>>>(cur, wc[i], cbp[i], nxt,
                                                   B, L[i].Cin, L[i].Lin,
                                                   L[i].Cout, L[i].Lout, K, Kpad);
        bf16_t* t = cur; cur = nxt; nxt = t;
    }
    // conv4 output is in `cur` (= pong); repack into `nxt` (= ping) as hflat.
    bf16_t* hflat = nxt;
    repack_hflat_kernel<<<(B * 8192 + 255) / 256, 256, 0, stream>>>(cur, hflat, B);

    // --- rank-2 input-path precompute ---
    prep_u_kernel<<<8, 256, 0, stream>>>(w_ih, emb_w, emb_b, b_ih, b_hh,
                                         u0, u1, u2, bvec);

    // --- initial state projections: h_s (bf16 into hb), c_s (fp32 into cbuf) ---
    gemm_wmma_kernel<<<dim3(B / 128, 512 / 64), 256, 0, stream>>>(
        hflat, phb, ph_b, nullptr, hb, B, 512, 8192);
    gemm_wmma_kernel<<<dim3(B / 128, 512 / 64), 256, 0, stream>>>(
        hflat, pcb, pc_b, cbuf, nullptr, B, 512, 8192);

    // --- recurrence: one WMMA GEMM + one fused cell kernel per step ---
    float* out = (float*)d_out;
    for (int t = 0; t < num_steps; ++t) {
        gemm_wmma_kernel<<<dim3(B / 128, 2048 / 64), 256, 0, stream>>>(
            hb, whhb, nullptr, gates, nullptr, B, 2048, 512);
        lstm_cell_kernel<<<B, 512, 0, stream>>>(
            gates, u0, u1, u2, bvec, cbuf, hb, lf0_w, lf0_b, uv_w, uv_b,
            sc, out, t, num_steps, t > 0 ? 1 : 0);
    }
}
